// APANLayer_72009421685070
// MI455X (gfx1250) — matrix operand order repaired and checked by reading, not compiled, not run on hardware
//
#include <hip/hip_runtime.h>
#include <hip/hip_bf16.h>
#include <math.h>

// ---------------------------------------------------------------------------
// APAN layer for MI455X (gfx1250, wave32, WMMA, async global->LDS)
//   B=4, L=2048, D=1024, H=16, d=64  -> M = B*L = 8192 rows
//
//   Stage 0: fp32 -> split-bf16 (hi/lo) copies of X and W     (one pass)
//   Stage 1: static_out = x @ W^T  (async-LDS double-buffered WMMA GEMM)
//   Stage 2: gated delta-rule scan (wave-synchronous, 1 wave per (b,h))
//   Stage 3: layernorm(static+o_dyn)
// ---------------------------------------------------------------------------

typedef __attribute__((ext_vector_type(16))) __bf16 v16bf;
typedef __attribute__((ext_vector_type(8)))  __bf16 v8bf;
typedef __attribute__((ext_vector_type(4)))  __bf16 bf16x4;
typedef __attribute__((ext_vector_type(8)))  float  v8f;

#define M_ROWS 8192
#define DMODEL 1024
#define TILE_M 128
#define TILE_N 128
#define TILE_K 32

union FragU {
    v8bf  h[2];   // two 16-byte LDS loads
    v16bf v;      // WMMA operand
};

// --- gfx1250 async global->LDS copy (ASYNCcnt) ------------------------------
__device__ __forceinline__ void async_load_b128(unsigned lds_off, unsigned goff,
                                                unsigned long long gbase) {
    asm volatile("global_load_async_to_lds_b128 %0, %1, %2 offset:0"
                 :: "v"(lds_off), "v"(goff), "s"(gbase)
                 : "memory");
}
__device__ __forceinline__ void wait_async0() {
    asm volatile("s_wait_asynccnt 0" ::: "memory");
}
// LDS byte offset of a __shared__ object (generic addr low 32 bits = LDS offset)
__device__ __forceinline__ unsigned lds_off_of(const void* p) {
    return (unsigned)(unsigned long long)p;
}

// -------------------------- Stage 0: split-bf16 convert ---------------------
// x = xh + xl, W = wh + wl;  GEMM later computes xh*wh + xh*wl + xl*wh.
__global__ __launch_bounds__(256) void apan_convert(
    const float* __restrict__ X, const float* __restrict__ Wm,
    __bf16* __restrict__ Xh, __bf16* __restrict__ Xl,
    __bf16* __restrict__ Wh, __bf16* __restrict__ Wl)
{
    const size_t i4  = (size_t)blockIdx.x * 256 + threadIdx.x;   // quad index
    const size_t NX4 = (size_t)M_ROWS * DMODEL / 4;              // X quads
    const bool  isX  = i4 < NX4;
    const size_t j   = isX ? i4 * 4 : (i4 - NX4) * 4;
    const float4 v   = isX ? *(const float4*)&X[j] : *(const float4*)&Wm[j];
    const float f[4] = {v.x, v.y, v.z, v.w};
    bf16x4 hi, lo;
    #pragma unroll
    for (int e = 0; e < 4; ++e) {
        hi[e] = (__bf16)f[e];
        lo[e] = (__bf16)(f[e] - (float)hi[e]);
    }
    if (isX) { *(bf16x4*)&Xh[j] = hi; *(bf16x4*)&Xl[j] = lo; }
    else     { *(bf16x4*)&Wh[j] = hi; *(bf16x4*)&Wl[j] = lo; }
}

// -------------------------- Stage 1: GEMM -----------------------------------
// C[m,n] = sum_k X[m,k] * W[n,k]; tiles DMA'd global->LDS asynchronously.
// Copy one 8KB tile (128 rows x 64B) : 512 b128 chunks / 256 threads = 2 each.
__device__ __forceinline__ void issue_tile_async(
    const __bf16* gbase, int grow0, int kt, unsigned lds_base, int tid)
{
    const unsigned long long gb = (unsigned long long)gbase;
    #pragma unroll
    for (int p = 0; p < 2; ++p) {
        const int chunk = tid + p * 256;        // 0..511
        const int row   = chunk >> 2;           // 0..127
        const int c16   = (chunk & 3) * 16;     // byte offset within row
        const unsigned goff = (unsigned)(((grow0 + row) * DMODEL + kt) * 2 + c16);
        async_load_b128(lds_base + (unsigned)(row * 64 + c16), goff, gb);
    }
}

__global__ __launch_bounds__(256) void apan_gemm_wmma(
    const __bf16* __restrict__ Xh, const __bf16* __restrict__ Xl,
    const __bf16* __restrict__ Wh, const __bf16* __restrict__ Wl,
    float* __restrict__ Y)
{
    __shared__ __align__(16) __bf16 Abuf[2][2][TILE_M * TILE_K]; // [buf][hi/lo]
    __shared__ __align__(16) __bf16 Bbuf[2][2][TILE_N * TILE_K];

    const int tid  = threadIdx.x;
    const int lane = tid & 31;
    const int wave = tid >> 5;
    const int wm   = (wave & 3) * 32;   // wave's M offset inside block tile
    const int wn   = (wave >> 2) * 64;  // wave's N offset inside block tile
    const int m0   = blockIdx.y * TILE_M;
    const int n0   = blockIdx.x * TILE_N;

    unsigned ldsA[2][2], ldsB[2][2];
    #pragma unroll
    for (int bu = 0; bu < 2; ++bu)
        #pragma unroll
        for (int s = 0; s < 2; ++s) {
            ldsA[bu][s] = lds_off_of(&Abuf[bu][s][0]);
            ldsB[bu][s] = lds_off_of(&Bbuf[bu][s][0]);
        }

    v8f acc[2][4];
    #pragma unroll
    for (int i = 0; i < 2; ++i)
        #pragma unroll
        for (int j = 0; j < 4; ++j)
            #pragma unroll
            for (int e = 0; e < 8; ++e) acc[i][j][e] = 0.0f;

    // prologue: DMA first k-tile into buffer 0
    issue_tile_async(Xh, m0, 0, ldsA[0][0], tid);
    issue_tile_async(Xl, m0, 0, ldsA[0][1], tid);
    issue_tile_async(Wh, n0, 0, ldsB[0][0], tid);
    issue_tile_async(Wl, n0, 0, ldsB[0][1], tid);

    for (int kt = 0; kt < DMODEL; kt += TILE_K) {
        const int buf = (kt >> 5) & 1;
        wait_async0();          // this wave's async copies done
        __syncthreads();        // all waves' copies done; prev compute retired
        if (kt + TILE_K < DMODEL) {
            const int nb = buf ^ 1;
            issue_tile_async(Xh, m0, kt + TILE_K, ldsA[nb][0], tid);
            issue_tile_async(Xl, m0, kt + TILE_K, ldsA[nb][1], tid);
            issue_tile_async(Wh, n0, kt + TILE_K, ldsB[nb][0], tid);
            issue_tile_async(Wl, n0, kt + TILE_K, ldsB[nb][1], tid);
        }

        const __bf16* Ah = Abuf[buf][0];
        const __bf16* Al = Abuf[buf][1];
        const __bf16* Bh = Bbuf[buf][0];
        const __bf16* Bl = Bbuf[buf][1];

        // A fragment (16x32 bf16): lane L -> row M=L%16, kbase=(L/16)*8,
        //   elems K = kbase..kbase+7 and kbase+16..kbase+23 (two b128 loads)
        FragU ahf[2], alf[2], bhf[4], blf[4];
        const int arow = lane & 15;
        const int akb  = (lane >> 4) * 8;
        #pragma unroll
        for (int ms = 0; ms < 2; ++ms) {
            const int r = wm + ms * 16 + arow;
            ahf[ms].h[0] = *(const v8bf*)&Ah[r * TILE_K + akb];
            ahf[ms].h[1] = *(const v8bf*)&Ah[r * TILE_K + akb + 16];
            alf[ms].h[0] = *(const v8bf*)&Al[r * TILE_K + akb];
            alf[ms].h[1] = *(const v8bf*)&Al[r * TILE_K + akb + 16];
        }
        // B fragment (32x16 bf16): lane L -> col N=L%16, kbase=(L/16)*8,
        //   elems K = kbase..kbase+7 and kbase+16..kbase+23 (same as A)
        const int bkb = (lane >> 4) * 8;
        #pragma unroll
        for (int ns = 0; ns < 4; ++ns) {
            const int c = wn + ns * 16 + (lane & 15);
            bhf[ns].h[0] = *(const v8bf*)&Bh[c * TILE_K + bkb];
            bhf[ns].h[1] = *(const v8bf*)&Bh[c * TILE_K + bkb + 16];
            blf[ns].h[0] = *(const v8bf*)&Bl[c * TILE_K + bkb];
            blf[ns].h[1] = *(const v8bf*)&Bl[c * TILE_K + bkb + 16];
        }

        #pragma unroll
        for (int ms = 0; ms < 2; ++ms)
            #pragma unroll
            for (int ns = 0; ns < 4; ++ns) {
                acc[ms][ns] = __builtin_amdgcn_wmma_f32_16x16x32_bf16(
                    false, ahf[ms].v, false, bhf[ns].v, (short)0, acc[ms][ns], false, false);
                acc[ms][ns] = __builtin_amdgcn_wmma_f32_16x16x32_bf16(
                    false, ahf[ms].v, false, blf[ns].v, (short)0, acc[ms][ns], false, false);
                acc[ms][ns] = __builtin_amdgcn_wmma_f32_16x16x32_bf16(
                    false, alf[ms].v, false, bhf[ns].v, (short)0, acc[ms][ns], false, false);
            }
    }

    // C/D layout: VGPR r -> lanes 0-15: (M=r, N=lane); lanes 16-31: (M=r+8, N=lane-16)
    #pragma unroll
    for (int ms = 0; ms < 2; ++ms)
        #pragma unroll
        for (int ns = 0; ns < 4; ++ns) {
            const int mb = m0 + wm + ms * 16 + ((lane >> 4) << 3);
            const int nc = n0 + wn + ns * 16 + (lane & 15);
            #pragma unroll
            for (int r = 0; r < 8; ++r)
                Y[(size_t)(mb + r) * DMODEL + nc] = acc[ms][ns][r];
        }
}

// -------------------------- Stage 2: gated delta-rule scan ------------------
// Wave-synchronous: one wave32 per (b,h); lane owns columns v=lane, lane+32.
// Per step: S=lam*S; err = v_t - k^T S; S += beta*k (x) err; o = q^T S.
// q=k, v_t = -static_out, beta = -(eta*(1-lam)/d). No barriers, no LDS.
__global__ __launch_bounds__(32) void apan_scan(
    const float* __restrict__ X, float* __restrict__ Y,
    const float* __restrict__ knw_g, const float* __restrict__ eta_raw,
    const float* __restrict__ lam_logit)
{
    const int lane = threadIdx.x;
    const int h    = blockIdx.x & 15;
    const int b    = blockIdx.x >> 4;
    const float lam  = 1.0f / (1.0f + __expf(-lam_logit[h]));
    const float beta = -(tanhf(eta_raw[h]) * (1.0f - lam) * (1.0f / 64.0f));
    const float kn0  = knw_g[lane];
    const float kn1  = knw_g[lane + 32];

    float S0[64], S1[64];
    #pragma unroll
    for (int r = 0; r < 64; ++r) { S0[r] = 0.0f; S1[r] = 0.0f; }

    size_t base = ((size_t)b * 2048) * DMODEL + h * 64 + lane;
    for (int t = 0; t < 2048; ++t, base += DMODEL) {
        if (t + 1 < 2048) {
            __builtin_prefetch(&X[base + DMODEL], 0, 0);
            __builtin_prefetch(&Y[base + DMODEL], 0, 0);
        }
        const float x0 = X[base], x1 = X[base + 32];
        const float s0 = Y[base], s1 = Y[base + 32];   // static_out

        // rmsnorm over the 64-wide head (eps = FLT_EPSILON), wave reduction
        float ss = fmaf(x0, x0, x1 * x1);
        #pragma unroll
        for (int off = 16; off > 0; off >>= 1) ss += __shfl_xor(ss, off, 32);
        const float rms = rsqrtf(ss * (1.0f / 64.0f) + 1.1920929e-07f);
        const float kv0 = x0 * rms * kn0;
        const float kv1 = x1 * rms * kn1;

        // err_v = v_v - lam * sum_r k_r S[r][v] ; v = -static
        float acc0[4] = {0.f, 0.f, 0.f, 0.f};
        float acc1[4] = {0.f, 0.f, 0.f, 0.f};
        #pragma unroll
        for (int r = 0; r < 64; ++r) {
            const unsigned kb = (r < 32)
                ? __builtin_amdgcn_readlane(__float_as_uint(kv0), r)
                : __builtin_amdgcn_readlane(__float_as_uint(kv1), r - 32);
            const float kr = __uint_as_float(kb);
            acc0[r & 3] = fmaf(kr, S0[r], acc0[r & 3]);
            acc1[r & 3] = fmaf(kr, S1[r], acc1[r & 3]);
        }
        const float err0 = -s0 - lam * ((acc0[0] + acc0[1]) + (acc0[2] + acc0[3]));
        const float err1 = -s1 - lam * ((acc1[0] + acc1[1]) + (acc1[2] + acc1[3]));
        const float c0 = beta * err0;
        const float c1 = beta * err1;

        // S = lam*S + c*k ;  o = k^T S_new
        float o0[4] = {0.f, 0.f, 0.f, 0.f};
        float o1[4] = {0.f, 0.f, 0.f, 0.f};
        #pragma unroll
        for (int r = 0; r < 64; ++r) {
            const unsigned kb = (r < 32)
                ? __builtin_amdgcn_readlane(__float_as_uint(kv0), r)
                : __builtin_amdgcn_readlane(__float_as_uint(kv1), r - 32);
            const float kr = __uint_as_float(kb);
            const float n0v = fmaf(c0, kr, lam * S0[r]);
            const float n1v = fmaf(c1, kr, lam * S1[r]);
            S0[r] = n0v;  S1[r] = n1v;
            o0[r & 3] = fmaf(kr, n0v, o0[r & 3]);
            o1[r & 3] = fmaf(kr, n1v, o1[r & 3]);
        }
        Y[base]      = s0 + ((o0[0] + o0[1]) + (o0[2] + o0[3]));
        Y[base + 32] = s1 + ((o1[0] + o1[1]) + (o1[2] + o1[3]));
    }
}

// -------------------------- Stage 3: layernorm ------------------------------
__global__ __launch_bounds__(256) void apan_layernorm(
    const float* __restrict__ Y, const float* __restrict__ w,
    const float* __restrict__ bvec, float* __restrict__ Out)
{
    const int row = blockIdx.x;
    const int tid = threadIdx.x;
    const float4 v = *(const float4*)&Y[(size_t)row * DMODEL + tid * 4];
    float s  = v.x + v.y + v.z + v.w;
    float sq = v.x * v.x + v.y * v.y + v.z * v.z + v.w * v.w;
    const int lane = tid & 31, wid = tid >> 5;
    #pragma unroll
    for (int off = 16; off > 0; off >>= 1) {
        s  += __shfl_xor(s,  off, 32);
        sq += __shfl_xor(sq, off, 32);
    }
    __shared__ float rs[8], rq[8];
    if (lane == 0) { rs[wid] = s; rq[wid] = sq; }
    __syncthreads();
    float ts = 0.0f, tq = 0.0f;
    #pragma unroll
    for (int i = 0; i < 8; ++i) { ts += rs[i]; tq += rq[i]; }
    const float mu  = ts * (1.0f / (float)DMODEL);
    const float var = tq * (1.0f / (float)DMODEL) - mu * mu;
    const float inv = rsqrtf(var + 1e-5f);
    const float4 wv = *(const float4*)&w[tid * 4];
    const float4 bv = *(const float4*)&bvec[tid * 4];
    float4 o;
    o.x = (v.x - mu) * inv * wv.x + bv.x;
    o.y = (v.y - mu) * inv * wv.y + bv.y;
    o.z = (v.z - mu) * inv * wv.z + bv.z;
    o.w = (v.w - mu) * inv * wv.w + bv.w;
    *(float4*)&Out[(size_t)row * DMODEL + tid * 4] = o;
}

// ---------------------------------------------------------------------------
extern "C" void kernel_launch(void* const* d_in, const int* in_sizes, int n_in,
                              void* d_out, int out_size, void* d_ws, size_t ws_size,
                              hipStream_t stream)
{
    const float* x         = (const float*)d_in[0]; // [4,2048,1024]
    const float* W         = (const float*)d_in[1]; // [1024,1024]
    const float* k_norm_w  = (const float*)d_in[2]; // [64]
    const float* o_norm_w  = (const float*)d_in[3]; // [1024]
    const float* o_norm_b  = (const float*)d_in[4]; // [1024]
    const float* eta_raw   = (const float*)d_in[5]; // [16]
    const float* lam_logit = (const float*)d_in[6]; // [16]
    float* out = (float*)d_out;

    // workspace layout: y (32MB) | Xh (16MB) | Xl (16MB) | Wh (2MB) | Wl (2MB)
    float*  y  = (float*)d_ws;
    __bf16* Xh = (__bf16*)((char*)d_ws + (size_t)M_ROWS * DMODEL * 4);
    __bf16* Xl = Xh + (size_t)M_ROWS * DMODEL;
    __bf16* Wh = Xl + (size_t)M_ROWS * DMODEL;
    __bf16* Wl = Wh + (size_t)DMODEL * DMODEL;

    (void)in_sizes; (void)n_in; (void)out_size; (void)ws_size;

    const int nQuads = (M_ROWS * DMODEL + DMODEL * DMODEL) / 4; // 2,359,296
    apan_convert<<<nQuads / 256, 256, 0, stream>>>(x, W, Xh, Xl, Wh, Wl);

    dim3 gGemm(DMODEL / TILE_N, M_ROWS / TILE_M);               // (8, 64)
    apan_gemm_wmma<<<gGemm, 256, 0, stream>>>(Xh, Xl, Wh, Wl, y);
    apan_scan<<<64, 32, 0, stream>>>(x, y, k_norm_w, eta_raw, lam_logit);
    apan_layernorm<<<M_ROWS, 256, 0, stream>>>(y, o_norm_w, o_norm_b, out);
}
